// MultiheadDiffAttn_81664508166591
// MI455X (gfx1250) — matrix operand order repaired and checked by reading, not compiled, not run on hardware
//
#include <hip/hip_runtime.h>

// Problem constants (from reference)
#define NB   4
#define NN   512
#define NE   512
#define NH   8
#define D2   64      // per-head value dim (2*HD)
#define NCH  16      // 2*H conv output channels
#define CIN  64      // u_proj in channels
#define LAMBDA_INIT 0.2f

typedef __attribute__((ext_vector_type(16))) __bf16 bf16x16;
typedef __attribute__((ext_vector_type(8)))  __bf16 bf16x8;
typedef __attribute__((ext_vector_type(8)))  float  f32x8;

__device__ __forceinline__ f32x8 wmma_bf16(bf16x16 a, bf16x16 b, f32x8 c) {
  // D = A(16x32 bf16) x B(32x16 bf16) + C(16x16 f32)
  return __builtin_amdgcn_wmma_f32_16x16x32_bf16(false, a, false, b, (short)0, c,
                                                 false, false);
}

// ---- Fragment loaders (CDNA5 wave32 WMMA layouts, ISA 7.12.2) ----
// A (16x32, 16-bit): lane L<16 -> row L, K {kh*8..+8, 16+kh*8..+8} with kh=L>>4
__device__ __forceinline__ bf16x16 frag_a_bf16(const __bf16* A, int lda, int m0, int k0) {
  const int lane = threadIdx.x & 31;
  const int row = lane & 15, kh = lane >> 4;
  const __bf16* p = A + (size_t)(m0 + row) * lda + k0 + kh * 8;
  bf16x8 lo = *(const bf16x8*)(p);
  bf16x8 hi = *(const bf16x8*)(p + 16);
  bf16x16 o;
#pragma unroll
  for (int i = 0; i < 8; ++i) { o[i] = lo[i]; o[i + 8] = hi[i]; }
  return o;
}
// B (32x16, 16-bit) from a transposed operand Bt[n][k] (contiguous in k):
// lane col = lane&15, K {kh*8..+8, 16+kh*8..+8}
__device__ __forceinline__ bf16x16 frag_bT_bf16(const __bf16* Bt, int ldb, int n0, int k0) {
  const int lane = threadIdx.x & 31;
  const int col = lane & 15, kh = lane >> 4;
  const __bf16* p = Bt + (size_t)(n0 + col) * ldb + k0 + kh * 8;
  bf16x8 lo = *(const bf16x8*)(p);
  bf16x8 hi = *(const bf16x8*)(p + 16);
  bf16x16 o;
#pragma unroll
  for (int i = 0; i < 8; ++i) { o[i] = lo[i]; o[i + 8] = hi[i]; }
  return o;
}
// Same, but sourced from f32 memory with on-the-fly cvt to bf16.
__device__ __forceinline__ bf16x16 frag_a_f32(const float* A, long lda, int m0, int k0) {
  const int lane = threadIdx.x & 31;
  const int row = lane & 15, kh = lane >> 4;
  const float* p = A + (size_t)(m0 + row) * lda + k0 + kh * 8;
  float4 x0 = *(const float4*)(p);
  float4 x1 = *(const float4*)(p + 4);
  float4 y0 = *(const float4*)(p + 16);
  float4 y1 = *(const float4*)(p + 20);
  bf16x16 o;
  o[0]=(__bf16)x0.x; o[1]=(__bf16)x0.y; o[2]=(__bf16)x0.z; o[3]=(__bf16)x0.w;
  o[4]=(__bf16)x1.x; o[5]=(__bf16)x1.y; o[6]=(__bf16)x1.z; o[7]=(__bf16)x1.w;
  o[8]=(__bf16)y0.x; o[9]=(__bf16)y0.y; o[10]=(__bf16)y0.z; o[11]=(__bf16)y0.w;
  o[12]=(__bf16)y1.x; o[13]=(__bf16)y1.y; o[14]=(__bf16)y1.z; o[15]=(__bf16)y1.w;
  return o;
}
__device__ __forceinline__ bf16x16 frag_bT_f32(const float* Bt, long ldb, int n0, int k0) {
  const int lane = threadIdx.x & 31;
  const int col = lane & 15, kh = lane >> 4;
  const float* p = Bt + (size_t)(n0 + col) * ldb + k0 + kh * 8;
  float4 x0 = *(const float4*)(p);
  float4 x1 = *(const float4*)(p + 4);
  float4 y0 = *(const float4*)(p + 16);
  float4 y1 = *(const float4*)(p + 20);
  bf16x16 o;
  o[0]=(__bf16)x0.x; o[1]=(__bf16)x0.y; o[2]=(__bf16)x0.z; o[3]=(__bf16)x0.w;
  o[4]=(__bf16)x1.x; o[5]=(__bf16)x1.y; o[6]=(__bf16)x1.z; o[7]=(__bf16)x1.w;
  o[8]=(__bf16)y0.x; o[9]=(__bf16)y0.y; o[10]=(__bf16)y0.z; o[11]=(__bf16)y0.w;
  o[12]=(__bf16)y1.x; o[13]=(__bf16)y1.y; o[14]=(__bf16)y1.z; o[15]=(__bf16)y1.w;
  return o;
}

// ---- Kernel: f32 -> bf16 convert ----
__global__ void k_cvt(const float* __restrict__ in, __bf16* __restrict__ out, int n) {
  int i = blockIdx.x * 256 + threadIdx.x;
  if (i < n) out[i] = (__bf16)in[i];
}

// ---- Kernel: lambda = exp(lq1.lk1) - exp(lq2.lk2) + 0.2 ----
__global__ void k_lambda(const float* __restrict__ lq1, const float* __restrict__ lk1,
                         const float* __restrict__ lq2, const float* __restrict__ lk2,
                         float* __restrict__ lamp) {
  __shared__ float r1[256], r2[256];
  float s1 = 0.f, s2 = 0.f;
  for (int i = threadIdx.x; i < NE; i += 256) {
    s1 += lq1[i] * lk1[i];
    s2 += lq2[i] * lk2[i];
  }
  r1[threadIdx.x] = s1; r2[threadIdx.x] = s2;
  __syncthreads();
  for (int off = 128; off; off >>= 1) {
    if (threadIdx.x < (unsigned)off) {
      r1[threadIdx.x] += r1[threadIdx.x + off];
      r2[threadIdx.x] += r2[threadIdx.x + off];
    }
    __syncthreads();
  }
  if (threadIdx.x == 0) lamp[0] = __expf(r1[0]) - __expf(r2[0]) + LAMBDA_INIT;
}

// ---- Kernel: v = x @ v_w^T, written transposed per head: vT[b][h][d][n] bf16 ----
// M = B*N = 2048, N = E = 512, K = 512. 4 waves/block, 1 16x16 tile per wave.
__global__ void k_gemm_v(const __bf16* __restrict__ xb, const __bf16* __restrict__ wb,
                         __bf16* __restrict__ vT) {
  const int tile = blockIdx.x * 4 + (threadIdx.x >> 5);
  const int mt = tile >> 5, nt = tile & 31;
  const int m0 = mt * 16, n0 = nt * 16;
  f32x8 c = {0.f,0.f,0.f,0.f,0.f,0.f,0.f,0.f};
  for (int k0 = 0; k0 < NE; k0 += 32)
    c = wmma_bf16(frag_a_bf16(xb, NE, m0, k0), frag_bT_bf16(wb, NE, n0, k0), c);
  const int lane = threadIdx.x & 31;
  const int col = lane & 15, kh = lane >> 4;
  const int f = n0 + col, h = f >> 6, d = f & 63;
#pragma unroll
  for (int r = 0; r < 8; ++r) {
    const int m = m0 + r + 8 * kh;
    const int b = m >> 9, n = m & 511;
    vT[(((size_t)b * NH + h) * D2 + d) * NN + n] = (__bf16)c[r];
  }
}

// ---- Kernel: fused u-proj (WMMA) + mask + dual softmax + diff ----
// One block per (b, query i). Softmax axis = u's first N axis (j).
__global__ void k_attnw(const float* __restrict__ u, const unsigned char* __restrict__ umask,
                        const float* __restrict__ uw, const float* __restrict__ ub,
                        const float* __restrict__ lamp, __bf16* __restrict__ diff) {
  __shared__ float ue[NN * NCH];    // 32 KB: [j][ch]
  __shared__ float red[256];
  __shared__ float mx[NCH];
  __shared__ float rs[NCH];

  const int b = blockIdx.x >> 9;
  const int i = blockIdx.x & 511;
  const int wave = threadIdx.x >> 5;
  const int lane = threadIdx.x & 31;
  const int arow = lane & 15, kh = lane >> 4, col = lane & 15;

  // B fragments (uproj_w is [16 x 64] row-major -> transposed-B layout)
  const bf16x16 bf0 = frag_bT_f32(uw, CIN, 0, 0);
  const bf16x16 bf1 = frag_bT_f32(uw, CIN, 0, 32);

  const float* ubase = u + (size_t)b * NN * NN * CIN + (size_t)i * CIN; // row j stride = NN*CIN
  const float bias = ub[col];

  for (int jt = wave; jt < 32; jt += 8) {
    const int j0 = jt * 16;
    if (jt + 8 < 32)
      __builtin_prefetch(ubase + (size_t)(j0 + 128 + arow) * (NN * CIN), 0, 1);
    f32x8 c = {0.f,0.f,0.f,0.f,0.f,0.f,0.f,0.f};
    c = wmma_bf16(frag_a_f32(ubase, (long)NN * CIN, j0, 0),  bf0, c);
    c = wmma_bf16(frag_a_f32(ubase, (long)NN * CIN, j0, 32), bf1, c);
#pragma unroll
    for (int r = 0; r < 8; ++r) {
      const int j = j0 + r + 8 * kh;
      float v = c[r] + bias;
      if (umask[((size_t)(b * NN + j) * NN + i) * NCH + col]) v = -__builtin_inff();
      ue[j * NCH + col] = v;
    }
  }
  __syncthreads();

  // softmax over j per channel (16 channels x 16 segments of 32 j)
  const int ch = threadIdx.x & 15;
  const int seg = threadIdx.x >> 4;
  float m = -__builtin_inff();
  for (int j = seg * 32; j < seg * 32 + 32; ++j) m = fmaxf(m, ue[j * NCH + ch]);
  red[threadIdx.x] = m;
  __syncthreads();
  if (threadIdx.x < NCH) {
    float mm = -__builtin_inff();
    for (int s = 0; s < 16; ++s) mm = fmaxf(mm, red[s * NCH + threadIdx.x]);
    mx[threadIdx.x] = mm;
  }
  __syncthreads();
  const float mmv = mx[ch];
  float psum = 0.f;
  for (int j = seg * 32; j < seg * 32 + 32; ++j) {
    const float e = __expf(ue[j * NCH + ch] - mmv);
    ue[j * NCH + ch] = e;
    psum += e;
  }
  red[threadIdx.x] = psum;
  __syncthreads();
  if (threadIdx.x < NCH) {
    float s = 0.f;
    for (int t = 0; t < 16; ++t) s += red[t * NCH + threadIdx.x];
    rs[threadIdx.x] = 1.f / s;
  }
  __syncthreads();

  const float lam = lamp[0];
  for (int idx = threadIdx.x; idx < NH * NN; idx += 256) {
    const int j = idx & 511, h = idx >> 9;
    const float p0 = ue[j * NCH + 2 * h]     * rs[2 * h];
    const float p1 = ue[j * NCH + 2 * h + 1] * rs[2 * h + 1];
    diff[(((size_t)b * NH + h) * NN + i) * NN + j] = (__bf16)(p0 - lam * p1);
  }
}

// ---- Kernel: attn = diff @ v per (b,h). M=512, N=64, K=512. f32 out. ----
__global__ void k_gemm_av(const __bf16* __restrict__ diff, const __bf16* __restrict__ vT,
                          float* __restrict__ attn) {
  const int tile = blockIdx.x * 4 + (threadIdx.x >> 5);   // 0..4095
  const int bh = tile >> 7;
  const int rem = tile & 127;
  const int mt = rem >> 2, nt = rem & 3;
  const int m0 = mt * 16, n0 = nt * 16;
  const __bf16* A  = diff + (size_t)bh * NN * NN;
  const __bf16* Bt = vT   + (size_t)bh * D2 * NN;
  f32x8 c = {0.f,0.f,0.f,0.f,0.f,0.f,0.f,0.f};
  for (int k0 = 0; k0 < NN; k0 += 32)
    c = wmma_bf16(frag_a_bf16(A, NN, m0, k0), frag_bT_bf16(Bt, NN, n0, k0), c);
  const int lane = threadIdx.x & 31;
  const int col = lane & 15, kh = lane >> 4;
#pragma unroll
  for (int r = 0; r < 8; ++r)
    attn[(size_t)bh * NN * D2 + (size_t)(m0 + r + 8 * kh) * D2 + n0 + col] = c[r];
}

// ---- Kernel: LayerNorm over d=64, write bf16 in [b, n, h*64+d] layout ----
__global__ void k_ln(const float* __restrict__ attn, const float* __restrict__ lnw,
                     const float* __restrict__ lnb, __bf16* __restrict__ xa) {
  const int row = blockIdx.x * 8 + (threadIdx.x >> 5);    // 0..16383 = (b,h,i)
  const int lane = threadIdx.x & 31;
  const float* a = attn + (size_t)row * D2;
  const float v0 = a[lane], v1 = a[lane + 32];
  float s = v0 + v1;
#pragma unroll
  for (int off = 16; off; off >>= 1) s += __shfl_xor(s, off, 32);
  const float mean = s * (1.f / 64.f);
  const float d0 = v0 - mean, d1 = v1 - mean;
  float q = d0 * d0 + d1 * d1;
#pragma unroll
  for (int off = 16; off; off >>= 1) q += __shfl_xor(q, off, 32);
  const float rstd = rsqrtf(q * (1.f / 64.f) + 1e-5f);
  const int bh = row >> 9, i = row & 511;
  const int b = bh >> 3, h = bh & 7;
  __bf16* o = xa + ((size_t)b * NN + i) * NE + h * D2;
  o[lane]      = (__bf16)(d0 * rstd * lnw[lane]      + lnb[lane]);
  o[lane + 32] = (__bf16)(d1 * rstd * lnw[lane + 32] + lnb[lane + 32]);
}

// ---- Kernel: out = attn_ln @ out_w^T. M=2048, N=512, K=512. f32 out. ----
__global__ void k_gemm_out(const __bf16* __restrict__ xa, const __bf16* __restrict__ ow,
                           float* __restrict__ out) {
  const int tile = blockIdx.x * 4 + (threadIdx.x >> 5);
  const int mt = tile >> 5, nt = tile & 31;
  const int m0 = mt * 16, n0 = nt * 16;
  f32x8 c = {0.f,0.f,0.f,0.f,0.f,0.f,0.f,0.f};
  for (int k0 = 0; k0 < NE; k0 += 32)
    c = wmma_bf16(frag_a_bf16(xa, NE, m0, k0), frag_bT_bf16(ow, NE, n0, k0), c);
  const int lane = threadIdx.x & 31;
  const int col = lane & 15, kh = lane >> 4;
#pragma unroll
  for (int r = 0; r < 8; ++r)
    out[(size_t)(m0 + r + 8 * kh) * NE + n0 + col] = c[r];
}

extern "C" void kernel_launch(void* const* d_in, const int* in_sizes, int n_in,
                              void* d_out, int out_size, void* d_ws, size_t ws_size,
                              hipStream_t stream) {
  (void)in_sizes; (void)n_in; (void)out_size; (void)ws_size;
  const float* x       = (const float*)d_in[0];
  const float* u       = (const float*)d_in[1];
  const unsigned char* umask = (const unsigned char*)d_in[2]; // jax bool = 1 byte
  const float* v_w     = (const float*)d_in[3];
  const float* out_w   = (const float*)d_in[4];
  const float* uproj_w = (const float*)d_in[5];
  const float* uproj_b = (const float*)d_in[6];
  const float* lq1     = (const float*)d_in[7];
  const float* lk1     = (const float*)d_in[8];
  const float* lq2     = (const float*)d_in[9];
  const float* lk2     = (const float*)d_in[10];
  const float* ln_w    = (const float*)d_in[11];
  const float* ln_b    = (const float*)d_in[12];
  float* out = (float*)d_out;

  char* ws = (char*)d_ws;
  __bf16* xb    = (__bf16*)(ws + 0);           //  2 MB  x as bf16 [2048,512]
  __bf16* vwb   = (__bf16*)(ws + 2097152);     // .5 MB  v_w bf16 [512,512]
  __bf16* owb   = (__bf16*)(ws + 2621440);     // .5 MB  out_w bf16
  __bf16* vT    = (__bf16*)(ws + 3145728);     //  2 MB  v transposed [B,H,64,N]
  __bf16* diff  = (__bf16*)(ws + 5242880);     // 16 MB  diff [B,H,N,N]
  float*  attn  = (float* )(ws + 22020096);    //  4 MB  attn [B,H,N,64]
  __bf16* xattn = (__bf16*)(ws + 26214400);    //  2 MB  LN'd attn [B,N,E]
  float*  lamp  = (float* )(ws + 28311552);    //  scalar lambda

  k_cvt<<<4096, 256, 0, stream>>>(x, xb, NB * NN * NE);
  k_cvt<<<1024, 256, 0, stream>>>(v_w, vwb, NE * NE);
  k_cvt<<<1024, 256, 0, stream>>>(out_w, owb, NE * NE);
  k_lambda<<<1, 256, 0, stream>>>(lq1, lk1, lq2, lk2, lamp);
  k_gemm_v<<<1024, 128, 0, stream>>>(xb, vwb, vT);
  k_attnw<<<NB * NN, 256, 0, stream>>>(u, umask, uproj_w, uproj_b, lamp, diff);
  k_gemm_av<<<1024, 128, 0, stream>>>(diff, vT, attn);
  k_ln<<<2048, 256, 0, stream>>>(attn, ln_w, ln_b, xattn);
  k_gemm_out<<<1024, 128, 0, stream>>>(xattn, owb, out);
}
